// EcgGCNModel_12292196402118
// MI455X (gfx1250) — matrix operand order repaired and checked by reading, not compiled, not run on hardware
//
#include <hip/hip_runtime.h>

typedef __attribute__((ext_vector_type(16))) _Float16 v16h;
typedef __attribute__((ext_vector_type(8)))  float    v8f;
typedef int v4i_b __attribute__((vector_size(16)));   // matches builtin's V4i

static __device__ __forceinline__ unsigned short f2h(float x) {
  _Float16 h = (_Float16)x;
  return __builtin_bit_cast(unsigned short, h);
}
static __device__ __forceinline__ unsigned int packh2(float a, float b) {
  return (unsigned int)f2h(a) | ((unsigned int)f2h(b) << 16);
}
static __device__ __forceinline__ float sigmoidf_(float x) {
  return 1.0f / (1.0f + __expf(-x));
}

union FragH { unsigned int u[8]; v16h v; };

// ---------------------------------------------------------------------------
// CDNA5 async global->LDS staging (ASYNCcnt path); sync fallback keeps the
// file compiling on toolchains lacking the builtin.
// ---------------------------------------------------------------------------
#if defined(__AMDGCN__) && __has_builtin(__builtin_amdgcn_global_load_async_to_lds_b128) && \
    __has_builtin(__builtin_amdgcn_s_wait_asynccnt)
#define USE_ASYNC_LDS 1
#else
#define USE_ASYNC_LDS 0
#endif

#if USE_ASYNC_LDS
static __device__ __forceinline__ void async_copy16(const void* g, void* l) {
  __builtin_amdgcn_global_load_async_to_lds_b128(
      (__attribute__((address_space(1))) v4i_b*)g,
      (__attribute__((address_space(3))) v4i_b*)l, 0, 0);
}
static __device__ __forceinline__ void async_wait0() {
  __builtin_amdgcn_s_wait_asynccnt(0);
}
#else
static __device__ __forceinline__ void async_copy16(const void* g, void* l) {
  *(v4i_b*)l = *(const v4i_b*)g;
}
static __device__ __forceinline__ void async_wait0() {}
#endif

// ---------------------------------------------------------------------------
// One-time producers: fp32 -> f16 conversion / transpose-conversion.
// ---------------------------------------------------------------------------
__global__ void cvt_f16x2(const float* __restrict__ X, unsigned int* __restrict__ Y)
{
  const long i = (long)blockIdx.x * blockDim.x + threadIdx.x;
  const float2 p = ((const float2*)X)[i];
  Y[i] = packh2(p.x, p.y);
}

// WH[n*K + k] = (f16) W[k*N + n]   (jax [K,N] -> [N,K] f16), 32x32 LDS tiles
__global__ void transpose_cvt(const float* __restrict__ W, _Float16* __restrict__ WH,
                              int K, int N)
{
  __shared__ _Float16 t[32][33];
  const int n0 = blockIdx.x * 32, k0 = blockIdx.y * 32;
  const int tx = threadIdx.x & 31, ty = threadIdx.x >> 5;   // 32 x 8
  #pragma unroll
  for (int i = 0; i < 32; i += 8)
    t[ty + i][tx] = (_Float16)W[(long)(k0 + ty + i) * N + n0 + tx];
  __syncthreads();
  #pragma unroll
  for (int i = 0; i < 32; i += 8)
    WH[(long)(n0 + ty + i) * K + k0 + tx] = t[tx][ty + i];
}

// ---------------------------------------------------------------------------
// Generic tiled WMMA GEMM: C[M,N](f32) = A[M,K](f16) * B[N,K](f16)^T + bias.
// 64x64 tile / 128 threads (4 wave32), K staged 32-wide through
// double-buffered f16 LDS via async-to-LDS; fragments are raw dword loads.
// AMODE: 0 = A row-major stride lda
//        1 = A rows gathered from XH in x[512,12,4,512] layout; row r=(b*4+w)*12+c
//        2 = A rows gathered from HSH[12][2048][512]; row r=n*12+c
// ---------------------------------------------------------------------------
template<int AMODE>
__global__ __launch_bounds__(128)
void gemm_wmma(const _Float16* __restrict__ A, long lda,
               const _Float16* __restrict__ Bm, long ldb,
               const float* __restrict__ bias,
               float* __restrict__ Cm, long ldc, int K)
{
  __shared__ _Float16 As[2][64][40];   // [m][k], 80B rows -> 16B-aligned chunks
  __shared__ _Float16 Bs[2][64][40];   // [n][k]

  const int tid = threadIdx.x;
  const int n0 = blockIdx.x * 64;
  const int m0 = blockIdx.y * 64;

  v8f acc[4];
  #pragma unroll
  for (int i = 0; i < 4; ++i) { v8f z = {0,0,0,0,0,0,0,0}; acc[i] = z; }

  const int nk = K >> 5;

  auto load_tiles = [&](int buf, int k0) {
    const int row = tid >> 1;                    // 0..63
    const _Float16* arow;
    if constexpr (AMODE == 0) {
      arow = A + (long)(m0 + row) * lda;
    } else if constexpr (AMODE == 1) {
      const long gr = m0 + row, c = gr % 12, n = gr / 12, b = n >> 2, w = n & 3;
      arow = A + ((b * 48 + c * 4 + w) << 9);
    } else {
      const long gr = m0 + row, c = gr % 12, n = gr / 12;
      arow = A + c * (2048L * 512) + (n << 9);
    }
    const _Float16* brow = Bm + (long)(n0 + row) * ldb;
    #pragma unroll
    for (int j = 0; j < 2; ++j) {
      const int c8 = ((tid & 1) * 2 + j) * 8;    // 0,8,16,24
      async_copy16(arow + k0 + c8, &As[buf][row][c8]);
      async_copy16(brow + k0 + c8, &Bs[buf][row][c8]);
    }
  };

  load_tiles(0, 0);
  async_wait0();
  __syncthreads();

  const int wv = tid >> 5, lane = tid & 31;
  const int lm = lane & 15, hi = lane >> 4;

  for (int kt = 0; kt < nk; ++kt) {
    const int cur = kt & 1;
    if (kt + 1 < nk) load_tiles(cur ^ 1, (kt + 1) << 5);

    // A fragment: 16-bit A 16x32 layout -> dword v holds K pair {2v,2v+1} (v<4)
    // or {8+2v,9+2v} (v>=4), +8 for the upper half-wave.
    FragH fa;
    {
      const unsigned short* ar = (const unsigned short*)As[cur][wv * 16 + lm];
      #pragma unroll
      for (int v = 0; v < 8; ++v) {
        const int kb = ((v < 4) ? (2 * v) : (8 + 2 * v)) + hi * 8;
        fa.u[v] = *(const unsigned int*)(ar + kb);
      }
    }
    #pragma unroll
    for (int ns = 0; ns < 4; ++ns) {
      FragH fb;   // B 32x16: same K order as A (K 0..7,16..23 low lanes; 8..15,24..31 high)
      const unsigned short* br = (const unsigned short*)Bs[cur][ns * 16 + lm];
      #pragma unroll
      for (int v = 0; v < 8; ++v)
        fb.u[v] = *(const unsigned int*)(br + ((v < 4) ? (2 * v) : (8 + 2 * v)) + hi * 8);
      acc[ns] = __builtin_amdgcn_wmma_f32_16x16x32_f16(
          false, fa.v, false, fb.v, (short)0, acc[ns], false, false);
    }
    async_wait0();
    __syncthreads();
  }

  // epilogue: f32 C/D layout -> row = v + 8*hi, col = lane&15
  #pragma unroll
  for (int ns = 0; ns < 4; ++ns) {
    const int col = n0 + ns * 16 + lm;
    const float bv = bias ? bias[col] : 0.0f;
    #pragma unroll
    for (int v = 0; v < 8; ++v) {
      const int row = m0 + wv * 16 + v + hi * 8;
      Cm[(long)row * ldc + col] = acc[ns][v] + bv;
    }
  }
}

// ---------------------------------------------------------------------------
// Fused GRU recurrence step: 64(n) x 64(f) tile of all three gate
// pre-activations (three accumulator sets vs the r/z/n row blocks of whh),
// gate math in the epilogue, h written as fp32 + f16.
// ---------------------------------------------------------------------------
__global__ __launch_bounds__(128)
void gru_step_fused(const _Float16* __restrict__ hprevH,  // [2048,512] f16 (A)
                    const float*    __restrict__ hprevF,  // [2048,512] f32
                    const _Float16* __restrict__ whhH,    // [1536,512] f16 (N,K)
                    const float*    __restrict__ XW,      // [24576,1536] (has b_ih)
                    const float*    __restrict__ bhh,     // [1536]
                    float*          __restrict__ houtF,   // [2048,512]
                    _Float16*       __restrict__ houtH,   // [2048,512]
                    int c)
{
  __shared__ _Float16 As[64][40];
  __shared__ _Float16 Bs[3][64][40];

  const int tid = threadIdx.x;
  const int f0 = blockIdx.x * 64;   // feature columns
  const int m0 = blockIdx.y * 64;   // n rows

  v8f acc[3][4];
  #pragma unroll
  for (int g = 0; g < 3; ++g)
    #pragma unroll
    for (int i = 0; i < 4; ++i) { v8f z = {0,0,0,0,0,0,0,0}; acc[g][i] = z; }

  const int wv = tid >> 5, lane = tid & 31;
  const int lm = lane & 15, hi = lane >> 4;

  for (int k0 = 0; k0 < 512; k0 += 32) {
    const int row = tid >> 1;
    #pragma unroll
    for (int j = 0; j < 2; ++j) {
      const int c8 = ((tid & 1) * 2 + j) * 8;
      async_copy16(hprevH + (long)(m0 + row) * 512 + k0 + c8, &As[row][c8]);
      #pragma unroll
      for (int g = 0; g < 3; ++g)
        async_copy16(whhH + (long)(g * 512 + f0 + row) * 512 + k0 + c8,
                     &Bs[g][row][c8]);
    }
    async_wait0();
    __syncthreads();

    FragH fa;
    {
      const unsigned short* ar = (const unsigned short*)As[wv * 16 + lm];
      #pragma unroll
      for (int v = 0; v < 8; ++v) {
        const int kb = ((v < 4) ? (2 * v) : (8 + 2 * v)) + hi * 8;
        fa.u[v] = *(const unsigned int*)(ar + kb);
      }
    }
    #pragma unroll
    for (int g = 0; g < 3; ++g) {
      #pragma unroll
      for (int ns = 0; ns < 4; ++ns) {
        FragH fb;
        const unsigned short* br = (const unsigned short*)Bs[g][ns * 16 + lm];
        #pragma unroll
        for (int v = 0; v < 8; ++v)
          fb.u[v] = *(const unsigned int*)(br + ((v < 4) ? (2 * v) : (8 + 2 * v)) + hi * 8);
        acc[g][ns] = __builtin_amdgcn_wmma_f32_16x16x32_f16(
            false, fa.v, false, fb.v, (short)0, acc[g][ns], false, false);
      }
    }
    __syncthreads();
  }

  // gate epilogue
  #pragma unroll
  for (int ns = 0; ns < 4; ++ns) {
    const int f = f0 + ns * 16 + lm;
    const float br_ = bhh[f], bz_ = bhh[512 + f], bn_ = bhh[1024 + f];
    #pragma unroll
    for (int v = 0; v < 8; ++v) {
      const int n = m0 + wv * 16 + v + hi * 8;
      const long rxw = ((long)n * 12 + c) * 1536 + f;
      const float xr = XW[rxw], xz = XW[rxw + 512], xn = XW[rxw + 1024];
      const float hp = hprevF[(long)n * 512 + f];
      const float rr = sigmoidf_(xr + acc[0][ns][v] + br_);
      const float zz = sigmoidf_(xz + acc[1][ns][v] + bz_);
      const float t  = __expf(-2.0f * (xn + rr * (acc[2][ns][v] + bn_)));
      const float nn = (1.0f - t) / (1.0f + t);
      const float h  = (1.0f - zz) * nn + zz * hp;
      houtF[(long)n * 512 + f] = h;
      houtH[(long)n * 512 + f] = (_Float16)h;
    }
  }
}

// GRU step 0: h_prev = 0, so hg = b_hh; pure elementwise.
__global__ void gru_init(const float* __restrict__ XW, const float* __restrict__ bhh,
                         float* __restrict__ houtF, _Float16* __restrict__ houtH)
{
  const long idx = (long)blockIdx.x * blockDim.x + threadIdx.x;  // < 2048*512
  const long n = idx >> 9;
  const int  f = (int)(idx & 511);
  const float* xw = XW + (n * 12) * 1536 + f;
  const float xr = xw[0], xz = xw[512], xn = xw[1024];
  const float rr = sigmoidf_(xr + bhh[f]);
  const float zz = sigmoidf_(xz + bhh[512 + f]);
  const float t  = __expf(-2.0f * (xn + rr * bhh[1024 + f]));
  const float nn = (1.0f - t) / (1.0f + t);
  const float h  = (1.0f - zz) * nn;
  houtF[idx] = h;
  houtH[idx] = (_Float16)h;
}

// ---------------------------------------------------------------------------
// Graph learning, stage 1: y2[b,c] = relu(sum_f relu(sum_w HS*c0w + b0)*c1w + b1)
// ---------------------------------------------------------------------------
__global__ void graph_y2(const float* __restrict__ HS, const float* __restrict__ c0w,
                         const float* __restrict__ c0b, const float* __restrict__ c1w,
                         const float* __restrict__ c1b, float* __restrict__ Y2)
{
  __shared__ float red[64];
  const int bc = blockIdx.x;
  const int b = bc / 12, c = bc % 12;
  const float w0 = c0w[0], w1 = c0w[1], w2 = c0w[2], w3 = c0w[3], b0 = c0b[0];
  const float* base = HS + ((long)c * 2048 + (long)b * 4) * 512;
  float p = 0.0f;
  for (int f = threadIdx.x; f < 512; f += 64) {
    float y = base[f] * w0 + base[512 + f] * w1 + base[1024 + f] * w2 + base[1536 + f] * w3 + b0;
    p += fmaxf(y, 0.0f) * c1w[f];
  }
  red[threadIdx.x] = p;
  __syncthreads();
  for (int s = 32; s > 0; s >>= 1) {
    if (threadIdx.x < s) red[threadIdx.x] += red[threadIdx.x + s];
    __syncthreads();
  }
  if (threadIdx.x == 0) Y2[bc] = fmaxf(red[0] + c1b[0], 0.0f);
}

// ---------------------------------------------------------------------------
// Graph learning, stage 2 (single block): adjacency -> normalized Laplacian ->
// Chebyshev stack [L0=0, L1=lap, L2=2*lap^2, L3=2*lap@L2 - lap]
// ---------------------------------------------------------------------------
__global__ void graph_cheb(const float* __restrict__ Y2, const float* __restrict__ c2w,
                           const float* __restrict__ c2b, float* __restrict__ CHB)
{
  __shared__ float a[144], lap[144], l2[144], deg[12], dh[12];
  const int t = threadIdx.x;
  if (t < 144) {
    float s = 0.0f;
    const float bias = c2b[t];
    for (int b = 0; b < 512; ++b) {
      float acc = bias;
      const float* y = Y2 + b * 12;
      #pragma unroll
      for (int c = 0; c < 12; ++c) acc += y[c] * c2w[t * 12 + c];
      s += fmaxf(acc, 0.0f);
    }
    a[t] = s * (1.0f / 512.0f);
  }
  __syncthreads();
  if (t < 12) {
    float s = 0.0f;
    for (int j = 0; j < 12; ++j) s += fmaxf(a[t * 12 + j], 0.0f);
    deg[t] = s;
    dh[t]  = 1.0f / (sqrtf(s) + 1e-7f);
  }
  __syncthreads();
  if (t < 144) {
    const int i = t / 12, j = t % 12;
    const float adjs = 0.5f * (fmaxf(a[i * 12 + j], 0.0f) + fmaxf(a[j * 12 + i], 0.0f));
    lap[t] = dh[i] * (((i == j) ? deg[i] : 0.0f) - adjs) * dh[j];
  }
  __syncthreads();
  if (t < 144) {
    const int i = t / 12, j = t % 12;
    float s = 0.0f;
    for (int k = 0; k < 12; ++k) s += lap[i * 12 + k] * lap[k * 12 + j];
    l2[t] = 2.0f * s;
  }
  __syncthreads();
  if (t < 144) {
    const int i = t / 12, j = t % 12;
    float s = 0.0f;
    for (int k = 0; k < 12; ++k) s += lap[i * 12 + k] * l2[k * 12 + j];
    CHB[t] = 0.0f;
    CHB[144 + t] = lap[t];
    CHB[288 + t] = l2[t];
    CHB[432 + t] = 2.0f * s - lap[t];
  }
}

// ---------------------------------------------------------------------------
// Cheb apply + ReLU + residual.  One block = one (n, c) row of 512 features.
// RESGATHER=1: residual gathered from HS stack (layer 1); 0: direct (layer 2).
// OUTH (optional) receives an f16 copy for the next GEMM's A operand.
// ---------------------------------------------------------------------------
template<int RESGATHER>
__global__ void gcn_cheb_apply(const float* __restrict__ SUP, const float* __restrict__ CHB,
                               const float* __restrict__ gbias, const float* __restrict__ RES,
                               float* __restrict__ OUT, _Float16* __restrict__ OUTH)
{
  __shared__ float ch[144];
  const long r = blockIdx.x;
  const int  f = threadIdx.x;
  const int  cc = (int)(r % 12);
  const long n  = r / 12;
  const int  w  = (int)(n & 3);
  if (f < 144) ch[f] = CHB[w * 144 + f];
  __syncthreads();
  float s = 0.0f;
  const float* sb = SUP + (n * 12) * 512 + f;
  #pragma unroll
  for (int j = 0; j < 12; ++j) s += ch[cc * 12 + j] * sb[(long)j * 512];
  s = fmaxf(s + gbias[f], 0.0f);
  const float res = RESGATHER ? RES[((long)cc * 2048 + n) * 512 + f]
                              : RES[r * 512 + f];
  const float o = s + res;
  OUT[r * 512 + f] = o;
  if (OUTH) OUTH[r * 512 + f] = (_Float16)o;
}

// FEAT[b, c*512+f] = sum_w ( h0 + h2 ), written as f16 (feeds fc1 GEMM only)
__global__ void gcn_feat(const float* __restrict__ HS, const float* __restrict__ H2,
                         _Float16* __restrict__ FEATH)
{
  const int bcq = blockIdx.x;                 // b*12 + c
  const int b = bcq / 12, c = bcq % 12;
  const int f = threadIdx.x;
  float s = 0.0f;
  #pragma unroll
  for (int w = 0; w < 4; ++w) {
    const long n = (long)b * 4 + w;
    s += HS[((long)c * 2048 + n) * 512 + f];
    s += H2[(n * 12 + c) * 512 + f];
  }
  FEATH[(long)b * 6144 + c * 512 + f] = (_Float16)s;
}

// eval-mode BN after leaky-relu; dual fp32 + f16 output
__global__ void bn_leaky(const float* __restrict__ X, const float* __restrict__ g,
                         const float* __restrict__ be, const float* __restrict__ m,
                         const float* __restrict__ v, float* __restrict__ Y,
                         _Float16* __restrict__ YH)
{
  const long idx = (long)blockIdx.x * blockDim.x + threadIdx.x;
  const int col = (int)(idx & 511);
  const float x = X[idx];
  const float l = (x > 0.0f) ? x : 0.01f * x;
  const float o = (l - m[col]) * rsqrtf(v[col] + 1e-5f) * g[col] + be[col];
  Y[idx] = o;
  YH[idx] = (_Float16)o;
}

// final tiny GEMV: [512,512] x [512,4]
__global__ void fc3_kernel(const float* __restrict__ H, const float* __restrict__ W3,
                           const float* __restrict__ b3, float* __restrict__ out)
{
  const int t = blockIdx.x * blockDim.x + threadIdx.x;   // < 2048
  const int b = t >> 2, k = t & 3;
  float s = b3[k];
  const float* h = H + (long)b * 512;
  for (int f = 0; f < 512; ++f) s += h[f] * W3[f * 4 + k];
  out[t] = s;
}

// ---------------------------------------------------------------------------
extern "C" void kernel_launch(void* const* d_in, const int* in_sizes, int n_in,
                              void* d_out, int out_size, void* d_ws, size_t ws_size,
                              hipStream_t stream)
{
  (void)in_sizes; (void)n_in; (void)out_size; (void)ws_size;

  const float* x    = (const float*)d_in[0];
  const float* wih  = (const float*)d_in[1];
  const float* whh  = (const float*)d_in[2];
  const float* bih  = (const float*)d_in[3];
  const float* bhh  = (const float*)d_in[4];
  const float* c0w  = (const float*)d_in[5];
  const float* c0b  = (const float*)d_in[6];
  const float* c1w  = (const float*)d_in[7];
  const float* c1b  = (const float*)d_in[8];
  const float* c2w  = (const float*)d_in[9];
  const float* c2b  = (const float*)d_in[10];
  const float* gcnw = (const float*)d_in[11];
  const float* gcnb = (const float*)d_in[12];
  const float* fc1w = (const float*)d_in[13];
  const float* fc1b = (const float*)d_in[14];
  const float* bn1g = (const float*)d_in[15];
  const float* bn1b = (const float*)d_in[16];
  const float* bn1m = (const float*)d_in[17];
  const float* bn1v = (const float*)d_in[18];
  const float* fc2w = (const float*)d_in[19];
  const float* fc2b = (const float*)d_in[20];
  const float* bn2g = (const float*)d_in[21];
  const float* bn2b = (const float*)d_in[22];
  const float* bn2m = (const float*)d_in[23];
  const float* bn2v = (const float*)d_in[24];
  const float* fc3w = (const float*)d_in[25];
  const float* fc3b = (const float*)d_in[26];
  float* out = (float*)d_out;

  // workspace layout: fp32 region then f16 region (f16 base stays 16B aligned)
  float* ws   = (float*)d_ws;
  float* XW   = ws;                       // 24576*1536 = 37,748,736 f
  float* SUP  = XW;                       // alias: used only after GRU done
  float* HS   = XW + 37748736L;           // 12*2048*512 = 12,582,912
  float* H1   = HS + 12582912L;
  float* H2   = H1 + 12582912L;
  float* T1   = H2 + 12582912L;           // 262,144
  float* T2   = T1 + 262144L;             // 262,144
  float* Y2   = T2 + 262144L;             //   6,144
  float* CHB  = Y2 + 6144L;               //     576
  _Float16* XH    = (_Float16*)(CHB + 576);   // 12,582,912 h
  _Float16* HSH   = XH + 12582912L;           // 12,582,912 h
  _Float16* H1H   = HSH + 12582912L;          // 12,582,912 h
  _Float16* FEATH = H1H + 12582912L;          //  3,145,728 h
  _Float16* WIHH  = FEATH + 3145728L;         //    786,432 h
  _Float16* WHHH  = WIHH + 786432L;           //    786,432 h
  _Float16* GW0H  = WHHH + 786432L;           //    262,144 h
  _Float16* GW1H  = GW0H + 262144L;           //    262,144 h
  _Float16* FC1WH = GW1H + 262144L;           //  3,145,728 h
  _Float16* FC2WH = FC1WH + 3145728L;         //    262,144 h
  _Float16* T2H   = FC2WH + 262144L;          //    262,144 h

  // 0) one-time fp32 -> f16 staging of GEMM operands
  cvt_f16x2<<<24576, 256, 0, stream>>>(x,   (unsigned int*)XH);    // 6,291,456 pairs
  cvt_f16x2<<<1536,  256, 0, stream>>>(wih, (unsigned int*)WIHH);  //   393,216 pairs
  cvt_f16x2<<<1536,  256, 0, stream>>>(whh, (unsigned int*)WHHH);
  transpose_cvt<<<dim3(16, 16),  256, 0, stream>>>(gcnw,               GW0H,  512,  512);
  transpose_cvt<<<dim3(16, 16),  256, 0, stream>>>(gcnw + 512 * 512,   GW1H,  512,  512);
  transpose_cvt<<<dim3(16, 192), 256, 0, stream>>>(fc1w,               FC1WH, 6144, 512);
  transpose_cvt<<<dim3(16, 16),  256, 0, stream>>>(fc2w,               FC2WH, 512,  512);

  // 1) GRU input projection: XW = xs @ wih^T + bih   (rows gathered from XH)
  gemm_wmma<1><<<dim3(1536 / 64, 24576 / 64), 128, 0, stream>>>(
      XH, 0, WIHH, 512, bih, XW, 1536, 512);

  // 2) GRU scan over 12 leads (step 0 elementwise, steps 1..11 fused GEMM+gates)
  gru_init<<<4096, 256, 0, stream>>>(XW, bhh, HS, HSH);
  for (int c = 1; c < 12; ++c) {
    const long off = (long)(c - 1) * 2048 * 512;
    gru_step_fused<<<dim3(512 / 64, 2048 / 64), 128, 0, stream>>>(
        HSH + off, HS + off, WHHH, XW, bhh,
        HS + off + 2048 * 512, HSH + off + 2048 * 512, c);
  }

  // 3) graph learning -> Chebyshev stack
  graph_y2<<<6144, 64, 0, stream>>>(HS, c0w, c0b, c1w, c1b, Y2);
  graph_cheb<<<1, 256, 0, stream>>>(Y2, c2w, c2b, CHB);

  // 4) GCN layer 1 (support rows gathered from HSH; residual = h0 gathered)
  gemm_wmma<2><<<dim3(512 / 64, 24576 / 64), 128, 0, stream>>>(
      HSH, 0, GW0H, 512, nullptr, SUP, 512, 512);
  gcn_cheb_apply<1><<<24576, 512, 0, stream>>>(SUP, CHB, gcnb, HS, H1, H1H);

  //    GCN layer 2 (residual = H1)
  gemm_wmma<0><<<dim3(512 / 64, 24576 / 64), 128, 0, stream>>>(
      H1H, 512, GW1H, 512, nullptr, SUP, 512, 512);
  gcn_cheb_apply<0><<<24576, 512, 0, stream>>>(SUP, CHB, gcnb + 512, H1, H2,
                                               (_Float16*)nullptr);

  // 5) FEAT[b, c*512+f] = sum_w (h0 + h2) -> f16 (feeds fc1 only)
  gcn_feat<<<6144, 512, 0, stream>>>(HS, H2, FEATH);

  // 6) classifier head
  gemm_wmma<0><<<dim3(512 / 64, 512 / 64), 128, 0, stream>>>(
      FEATH, 6144, FC1WH, 6144, fc1b, T1, 512, 6144);
  bn_leaky<<<1024, 256, 0, stream>>>(T1, bn1g, bn1b, bn1m, bn1v, T2, T2H);
  gemm_wmma<0><<<dim3(512 / 64, 512 / 64), 128, 0, stream>>>(
      T2H, 512, FC2WH, 512, fc2b, T1, 512, 512);
  bn_leaky<<<1024, 256, 0, stream>>>(T1, bn2g, bn2b, bn2m, bn2v, T2, T2H);
  fc3_kernel<<<8, 256, 0, stream>>>(T2, fc3w, fc3b, out);
}
